// RoPEMultiHeadAttention_88450556494143
// MI455X (gfx1250) — matrix operand order repaired and checked by reading, not compiled, not run on hardware
//
#include <hip/hip_runtime.h>
#include <stdint.h>

// ---------------------------------------------------------------------------
// Types for CDNA5 WMMA (wave32, v_wmma_f32_16x16x32_bf16)
// ---------------------------------------------------------------------------
typedef __attribute__((ext_vector_type(16))) __bf16 v16bf;
typedef __attribute__((ext_vector_type(8)))  __bf16 v8bf;
typedef __attribute__((ext_vector_type(8)))  float  v8f;

__device__ inline uint16_t f2bf_u(float f) {
    union { float f; uint32_t u; } x; x.f = f;
    uint32_t r = (x.u + 0x7FFFu + ((x.u >> 16) & 1u)) >> 16;  // RNE
    return (uint16_t)r;
}
__device__ inline __bf16 u2bf(uint16_t u) {
    union { uint16_t u; __bf16 b; } y; y.u = u; return y.b;
}

__device__ inline v8f wmma_bf16(v16bf a, v16bf b, v8f c) {
    // (neg_a, A, neg_b, B, c_mod, C, reuse_a, reuse_b)
    return __builtin_amdgcn_wmma_f32_16x16x32_bf16(false, a, false, b,
                                                   (short)0, c, false, false);
}

// ---------------------------------------------------------------------------
// Fragment loads matching CDNA5 16-bit VGPR layouts (CDNA5 ISA 7.12.2)
// A (16x32, MxK): lane<16 holds M=lane, K = [kk+hi*8 .. +7] U [kk+16+hi*8 .. +7]
// B (32x16, KxN): lane holds N=lane&15, K = [kk+hi*8 .. +7] U [kk+16+hi*8 .. +7]
// C (16x16 f32):  element r of lane -> row r + hi*8, col lane&15
// ---------------------------------------------------------------------------
__device__ inline v16bf load_a_bf16(const uint16_t* __restrict__ A, int lda,
                                    int row0, int kk, int lane) {
    int m  = row0 + (lane & 15);
    int hi = lane >> 4;
    const uint16_t* base = A + (size_t)m * lda + kk + hi * 8;
    v8bf lo = *reinterpret_cast<const v8bf*>(base);
    v8bf hh = *reinterpret_cast<const v8bf*>(base + 16);
    v16bf r;
#pragma unroll
    for (int i = 0; i < 8; ++i) { r[i] = lo[i]; r[8 + i] = hh[i]; }
    return r;
}

__device__ inline v16bf load_a_f32(const float* __restrict__ A, int lda,
                                   int row0, int kk, int lane) {
    int m  = row0 + (lane & 15);
    int hi = lane >> 4;
    const float* base = A + (size_t)m * lda + kk + hi * 8;
    v8f lo = *reinterpret_cast<const v8f*>(base);
    v8f hh = *reinterpret_cast<const v8f*>(base + 16);
    v16bf r;
#pragma unroll
    for (int i = 0; i < 8; ++i) {
        r[i]     = u2bf(f2bf_u(lo[i]));
        r[8 + i] = u2bf(f2bf_u(hh[i]));
    }
    return r;
}

__device__ inline v16bf load_b_bf16(const uint16_t* __restrict__ W, int ldb,
                                    int col0, int kk, int lane) {
    int n  = col0 + (lane & 15);
    int ks = kk + (lane >> 4) * 8;
    const uint16_t* base = W + (size_t)n * ldb + ks;
    v8bf lo = *reinterpret_cast<const v8bf*>(base);
    v8bf hh = *reinterpret_cast<const v8bf*>(base + 16);
    v16bf r;
#pragma unroll
    for (int i = 0; i < 8; ++i) { r[i] = lo[i]; r[8 + i] = hh[i]; }
    return r;
}

// ---------------------------------------------------------------------------
// Elementwise fp32 -> bf16 conversion
// ---------------------------------------------------------------------------
__global__ __launch_bounds__(256) void f32_to_bf16_k(const float* __restrict__ in,
                                                     uint16_t* __restrict__ out,
                                                     int n) {
    int i = blockIdx.x * 256 + threadIdx.x;
    int stride = gridDim.x * 256;
    for (; i < n; i += stride) out[i] = f2bf_u(in[i]);
}

// ---------------------------------------------------------------------------
// GEMM: C(MxN) = A(MxK) @ B(NxK)^T, bf16 operands, fp32 out.
// Block 256 threads = 8 waves (4x2), each wave a 32x32 tile -> block 128x64.
// ---------------------------------------------------------------------------
__global__ __launch_bounds__(256) void gemm_qkv(const uint16_t* __restrict__ A,
                                                const uint16_t* __restrict__ B,
                                                float* __restrict__ C,
                                                int N, int K) {
    int lane = threadIdx.x & 31, wave = threadIdx.x >> 5;
    int rowBase = blockIdx.x * 128 + (wave & 3) * 32;
    int colBase = blockIdx.y * 64  + (wave >> 2) * 32;
    v8f acc[2][2];
#pragma unroll
    for (int t = 0; t < 2; ++t)
#pragma unroll
        for (int u = 0; u < 2; ++u) acc[t][u] = (v8f)0.0f;

    for (int kk = 0; kk < K; kk += 32) {
        v16bf a0 = load_a_bf16(A, K, rowBase,      kk, lane);
        v16bf a1 = load_a_bf16(A, K, rowBase + 16, kk, lane);
        v16bf b0 = load_b_bf16(B, K, colBase,      kk, lane);
        v16bf b1 = load_b_bf16(B, K, colBase + 16, kk, lane);
        acc[0][0] = wmma_bf16(a0, b0, acc[0][0]);
        acc[0][1] = wmma_bf16(a0, b1, acc[0][1]);
        acc[1][0] = wmma_bf16(a1, b0, acc[1][0]);
        acc[1][1] = wmma_bf16(a1, b1, acc[1][1]);
    }
    int hi = lane >> 4, nl = lane & 15;
#pragma unroll
    for (int t = 0; t < 2; ++t)
#pragma unroll
        for (int u = 0; u < 2; ++u)
#pragma unroll
            for (int r = 0; r < 8; ++r) {
                int m = rowBase + t * 16 + hi * 8 + r;
                int n = colBase + u * 16 + nl;
                C[(size_t)m * N + n] = acc[t][u][r];
            }
}

// Batched scores: per (b,h): S x S = Q(SxD) @ K(SxD)^T * scale + mask
__global__ __launch_bounds__(256) void gemm_scores(const uint16_t* __restrict__ Q,
                                                   const uint16_t* __restrict__ Kb,
                                                   const float* __restrict__ mask,
                                                   float* __restrict__ attn,
                                                   int S, int D, float scale) {
    int z = blockIdx.z;
    const uint16_t* Az = Q  + (size_t)z * S * D;
    const uint16_t* Bz = Kb + (size_t)z * S * D;
    float*          Cz = attn + (size_t)z * S * S;

    int lane = threadIdx.x & 31, wave = threadIdx.x >> 5;
    int rowBase = blockIdx.x * 128 + (wave & 3) * 32;
    int colBase = blockIdx.y * 64  + (wave >> 2) * 32;
    v8f acc[2][2];
#pragma unroll
    for (int t = 0; t < 2; ++t)
#pragma unroll
        for (int u = 0; u < 2; ++u) acc[t][u] = (v8f)0.0f;

    for (int kk = 0; kk < D; kk += 32) {
        v16bf a0 = load_a_bf16(Az, D, rowBase,      kk, lane);
        v16bf a1 = load_a_bf16(Az, D, rowBase + 16, kk, lane);
        v16bf b0 = load_b_bf16(Bz, D, colBase,      kk, lane);
        v16bf b1 = load_b_bf16(Bz, D, colBase + 16, kk, lane);
        acc[0][0] = wmma_bf16(a0, b0, acc[0][0]);
        acc[0][1] = wmma_bf16(a0, b1, acc[0][1]);
        acc[1][0] = wmma_bf16(a1, b0, acc[1][0]);
        acc[1][1] = wmma_bf16(a1, b1, acc[1][1]);
    }
    int hi = lane >> 4, nl = lane & 15;
#pragma unroll
    for (int t = 0; t < 2; ++t)
#pragma unroll
        for (int u = 0; u < 2; ++u)
#pragma unroll
            for (int r = 0; r < 8; ++r) {
                int m = rowBase + t * 16 + hi * 8 + r;
                int n = colBase + u * 16 + nl;
                Cz[(size_t)m * S + n] = acc[t][u][r] * scale + mask[(size_t)m * S + n];
            }
}

// Batched PV: per (b,h): ctx(SxD) = attn(SxS, fp32) @ Vt(DxS)^T, write bf16 in (B,S,E)
__global__ __launch_bounds__(256) void gemm_pv(const float* __restrict__ attn,
                                               const uint16_t* __restrict__ Vt,
                                               uint16_t* __restrict__ ctx,
                                               int S, int D, int H, int E) {
    int z = blockIdx.z;
    int b = z / H, h = z % H;
    const float*    Az = attn + (size_t)z * S * S;
    const uint16_t* Bz = Vt   + (size_t)z * D * S;

    int lane = threadIdx.x & 31, wave = threadIdx.x >> 5;
    int rowBase = blockIdx.x * 128 + (wave & 3) * 32;
    int colBase = blockIdx.y * 64  + (wave >> 2) * 32;
    v8f acc[2][2];
#pragma unroll
    for (int t = 0; t < 2; ++t)
#pragma unroll
        for (int u = 0; u < 2; ++u) acc[t][u] = (v8f)0.0f;

    for (int kk = 0; kk < S; kk += 32) {
        v16bf a0 = load_a_f32(Az, S, rowBase,      kk, lane);
        v16bf a1 = load_a_f32(Az, S, rowBase + 16, kk, lane);
        v16bf b0 = load_b_bf16(Bz, S, colBase,      kk, lane);
        v16bf b1 = load_b_bf16(Bz, S, colBase + 16, kk, lane);
        acc[0][0] = wmma_bf16(a0, b0, acc[0][0]);
        acc[0][1] = wmma_bf16(a0, b1, acc[0][1]);
        acc[1][0] = wmma_bf16(a1, b0, acc[1][0]);
        acc[1][1] = wmma_bf16(a1, b1, acc[1][1]);
    }
    int hi = lane >> 4, nl = lane & 15;
#pragma unroll
    for (int t = 0; t < 2; ++t)
#pragma unroll
        for (int u = 0; u < 2; ++u)
#pragma unroll
            for (int r = 0; r < 8; ++r) {
                int m = rowBase + t * 16 + hi * 8 + r;     // query position
                int n = colBase + u * 16 + nl;             // head dim
                ctx[((size_t)b * S + m) * E + h * D + n] = f2bf_u(acc[t][u][r]);
            }
}

// Out projection: C = A(MxK) @ W(NxK)^T + bias
__global__ __launch_bounds__(256) void gemm_out(const uint16_t* __restrict__ A,
                                                const uint16_t* __restrict__ B,
                                                const float* __restrict__ bias,
                                                float* __restrict__ C,
                                                int N, int K) {
    int lane = threadIdx.x & 31, wave = threadIdx.x >> 5;
    int rowBase = blockIdx.x * 128 + (wave & 3) * 32;
    int colBase = blockIdx.y * 64  + (wave >> 2) * 32;
    v8f acc[2][2];
#pragma unroll
    for (int t = 0; t < 2; ++t)
#pragma unroll
        for (int u = 0; u < 2; ++u) acc[t][u] = (v8f)0.0f;

    for (int kk = 0; kk < K; kk += 32) {
        v16bf a0 = load_a_bf16(A, K, rowBase,      kk, lane);
        v16bf a1 = load_a_bf16(A, K, rowBase + 16, kk, lane);
        v16bf b0 = load_b_bf16(B, K, colBase,      kk, lane);
        v16bf b1 = load_b_bf16(B, K, colBase + 16, kk, lane);
        acc[0][0] = wmma_bf16(a0, b0, acc[0][0]);
        acc[0][1] = wmma_bf16(a0, b1, acc[0][1]);
        acc[1][0] = wmma_bf16(a1, b0, acc[1][0]);
        acc[1][1] = wmma_bf16(a1, b1, acc[1][1]);
    }
    int hi = lane >> 4, nl = lane & 15;
#pragma unroll
    for (int t = 0; t < 2; ++t)
#pragma unroll
        for (int u = 0; u < 2; ++u)
#pragma unroll
            for (int r = 0; r < 8; ++r) {
                int m = rowBase + t * 16 + hi * 8 + r;
                int n = colBase + u * 16 + nl;
                C[(size_t)m * N + n] = acc[t][u][r] + bias[n];
            }
}

// ---------------------------------------------------------------------------
// RoPE + repack: qkv fp32 (B*S, 3E) -> qb,kb bf16 (B,H,S,D); v -> Vt bf16 (B,H,D,S)
// ---------------------------------------------------------------------------
__global__ __launch_bounds__(256) void rope_pack(const float* __restrict__ qkv,
                                                 const float* __restrict__ cosT,
                                                 const float* __restrict__ sinT,
                                                 uint16_t* __restrict__ qb,
                                                 uint16_t* __restrict__ kb,
                                                 uint16_t* __restrict__ vt,
                                                 int S, int H, int D) {
    int d  = threadIdx.x & 63;            // D == 64
    int si = threadIdx.x >> 6;
    int s  = blockIdx.x * 4 + si;
    int h  = blockIdx.y, b = blockIdx.z;
    int E3 = 3 * H * D;

    size_t m = (size_t)b * S + s;
    const float* base = qkv + m * E3;
    int i2 = d & ~1;                       // even index of the pair
    float c  = cosT[(size_t)s * D + i2];
    float sn = sinT[(size_t)s * D + i2];

    int qoff = h * D;
    float x1 = base[qoff + i2], x2 = base[qoff + i2 + 1];
    float qv = (d & 1) ? (x1 * sn + x2 * c) : (x1 * c - x2 * sn);

    int koff = (H + h) * D;
    float y1 = base[koff + i2], y2 = base[koff + i2 + 1];
    float kv = (d & 1) ? (y1 * sn + y2 * c) : (y1 * c - y2 * sn);

    float vv = base[(2 * H + h) * D + d];

    size_t bh = (size_t)b * H + h;
    qb[(bh * S + s) * D + d] = f2bf_u(qv);
    kb[(bh * S + s) * D + d] = f2bf_u(kv);
    vt[(bh * D + d) * S + s] = f2bf_u(vv);
}

// ---------------------------------------------------------------------------
// Row softmax in place (rows of length S)
// ---------------------------------------------------------------------------
__global__ __launch_bounds__(256) void softmax_rows(float* __restrict__ attn, int S) {
    __shared__ float red[256];
    float* row = attn + (size_t)blockIdx.x * S;
    int t = threadIdx.x;

    float mx = -3.4e38f;
    for (int i = t; i < S; i += 256) mx = fmaxf(mx, row[i]);
    red[t] = mx; __syncthreads();
    for (int off = 128; off > 0; off >>= 1) {
        if (t < off) red[t] = fmaxf(red[t], red[t + off]);
        __syncthreads();
    }
    mx = red[0]; __syncthreads();

    float sum = 0.0f;
    for (int i = t; i < S; i += 256) {
        float e = __expf(row[i] - mx);
        row[i] = e;
        sum += e;
    }
    red[t] = sum; __syncthreads();
    for (int off = 128; off > 0; off >>= 1) {
        if (t < off) red[t] += red[t + off];
        __syncthreads();
    }
    float inv = 1.0f / red[0];
    for (int i = t; i < S; i += 256) row[i] *= inv;
}

// ---------------------------------------------------------------------------
// Host launcher
// ---------------------------------------------------------------------------
extern "C" void kernel_launch(void* const* d_in, const int* in_sizes, int n_in,
                              void* d_out, int out_size, void* d_ws, size_t ws_size,
                              hipStream_t stream) {
    constexpr int B = 4, S = 1024, E = 1024, H = 16, D = 64, E3 = 3072;

    const float* x     = (const float*)d_in[0];
    const float* mask  = (const float*)d_in[1];
    const float* cosT  = (const float*)d_in[2];
    const float* sinT  = (const float*)d_in[3];
    const float* w_qkv = (const float*)d_in[4];
    const float* w_out = (const float*)d_in[5];
    const float* b_out = (const float*)d_in[6];

    float* out  = (float*)d_out;
    float* attn = out + (size_t)B * S * E;   // second output, flat after first

    // Workspace carve-up (256B aligned)
    char* ws = (char*)d_ws;
    auto carve = [&](size_t bytes) -> void* {
        void* p = ws;
        ws += (bytes + 255) & ~(size_t)255;
        return p;
    };
    float*    qkv   = (float*)   carve((size_t)B * S * E3 * sizeof(float));
    uint16_t* xb    = (uint16_t*)carve((size_t)B * S * E  * 2);
    uint16_t* wqkvb = (uint16_t*)carve((size_t)E3 * E     * 2);
    uint16_t* woutb = (uint16_t*)carve((size_t)E * E      * 2);
    uint16_t* qb    = (uint16_t*)carve((size_t)B * H * S * D * 2);
    uint16_t* kb    = (uint16_t*)carve((size_t)B * H * S * D * 2);
    uint16_t* vtb   = (uint16_t*)carve((size_t)B * H * D * S * 2);
    uint16_t* ctxb  = (uint16_t*)carve((size_t)B * S * E  * 2);

    // 1) fp32 -> bf16 operand conversion
    {
        int n1 = B * S * E, n2 = E3 * E, n3 = E * E;
        f32_to_bf16_k<<<(n1 + 255) / 256, 256, 0, stream>>>(x, xb, n1);
        f32_to_bf16_k<<<(n2 + 255) / 256, 256, 0, stream>>>(w_qkv, wqkvb, n2);
        f32_to_bf16_k<<<(n3 + 255) / 256, 256, 0, stream>>>(w_out, woutb, n3);
    }

    // 2) QKV projection: (4096 x 3072) = xb (4096x1024) @ wqkvb(3072x1024)^T
    gemm_qkv<<<dim3(B * S / 128, E3 / 64), 256, 0, stream>>>(xb, wqkvb, qkv, E3, E);

    // 3) RoPE + repack (q,k rotated; v transposed per head)
    rope_pack<<<dim3(S / 4, H, B), 256, 0, stream>>>(qkv, cosT, sinT, qb, kb, vtb, S, H, D);

    // 4) Scores: per (b,h) S x S, scale 1/sqrt(D), + mask -> attn region of d_out
    gemm_scores<<<dim3(S / 128, S / 64, B * H), 256, 0, stream>>>(
        qb, kb, mask, attn, S, D, 0.125f);

    // 5) Softmax rows in place
    softmax_rows<<<B * H * S, 256, 0, stream>>>(attn, S);

    // 6) PV: ctx bf16 written directly in (B,S,E) layout
    gemm_pv<<<dim3(S / 128, D / 64, B * H), 256, 0, stream>>>(attn, vtb, ctxb, S, D, H, E);

    // 7) Output projection + bias -> first region of d_out
    gemm_out<<<dim3(B * S / 128, E / 64), 256, 0, stream>>>(ctxb, woutb, b_out, out, E, E);
}
